// DKBATNet_4990751998391
// MI455X (gfx1250) — matrix operand order repaired and checked by reading, compiled, not run on hardware
//
#include <hip/hip_runtime.h>
#include <hip/hip_bf16.h>
#include <math.h>

// ---------------------------------------------------------------------------
// Types for CDNA5 WMMA (wave32): bf16 A/B fragments, f32 accumulator.
// ---------------------------------------------------------------------------
typedef __attribute__((ext_vector_type(16))) __bf16 v16bf;
typedef __attribute__((ext_vector_type(8)))  float  v8f;

#define DEV_ATOMIC_ADD(p, v) \
    __hip_atomic_fetch_add((p), (v), __ATOMIC_RELAXED, __HIP_MEMORY_SCOPE_AGENT)

__device__ __forceinline__ float wave_reduce_sum(float v) {
#pragma unroll
    for (int m = 16; m >= 1; m >>= 1) v += __shfl_xor(v, m, 32);
    return v;
}

// ---------------------------------------------------------------------------
// GEMM: O[r, 0:64] = X[r, 0:64] @ W[:, kofs:kofs+64].T  (+ optional bias)
// X: [R,64] f32 row-major.  W: [64, ldw] f32 row-major.  O: [R,64] f32.
// One wave computes a 16x64 tile: 4 N-tiles x 2 K-steps = 8 WMMAs, f32 accum.
// R is always a multiple of 16 here (N=50000=16*3125, M=256), so wave-level
// bounds guard keeps EXEC all-ones inside the WMMA path (ISA requirement).
// ---------------------------------------------------------------------------
__global__ __launch_bounds__(256)
void gemm_x64(const float* __restrict__ X, int R,
              const float* __restrict__ W, int ldw, int kofs,
              const float* __restrict__ bias,
              float* __restrict__ O)
{
    const int lane = threadIdx.x & 31;
    const int wave = threadIdx.x >> 5;
    const int m0   = (blockIdx.x * 8 + wave) * 16;
    if (m0 >= R) return;                 // uniform per wave
    const int lh = lane >> 4;            // lane half: 0 (lanes 0-15) / 1 (16-31)
    const int ll = lane & 15;

    v8f c[4];
#pragma unroll
    for (int jt = 0; jt < 4; ++jt)
#pragma unroll
        for (int i = 0; i < 8; ++i) c[jt][i] = 0.0f;

#pragma unroll
    for (int ks = 0; ks < 2; ++ks) {
        const int kbase = ks * 32;
        // A fragment: 16x32 bf16, contiguous k order (same as B).
        // lanes 0-15: halves 0..15 -> K=kbase+0..15
        // lanes 16-31: halves 0..15 -> K=kbase+16..31
        const float* xr = X + (size_t)(m0 + ll) * 64 + kbase + (lh ? 16 : 0);
        v16bf a;
#pragma unroll
        for (int i = 0; i < 8; ++i) a[i]     = (__bf16)xr[i];
#pragma unroll
        for (int i = 0; i < 8; ++i) a[8 + i] = (__bf16)xr[8 + i];

#pragma unroll
        for (int jt = 0; jt < 4; ++jt) {
            // B fragment: 32x16 bf16, B[k][n] = W[jbase+n][kofs+k].
            // lanes 0-15 hold K=kbase+0..15, lanes 16-31 hold K=kbase+16..31.
            const float* wr = W + (size_t)(jt * 16 + ll) * ldw + kofs + kbase + (lh ? 16 : 0);
            v16bf b;
#pragma unroll
            for (int i = 0; i < 16; ++i) b[i] = (__bf16)wr[i];
            c[jt] = __builtin_amdgcn_wmma_f32_16x16x32_bf16(
                        false, a, false, b, (short)0, c[jt], false, false);
        }
    }
    // D layout: VGPR v, lanes 0-15 -> row m0+v, lanes 16-31 -> row m0+8+v; col = jbase + (lane&15)
#pragma unroll
    for (int jt = 0; jt < 4; ++jt) {
        const int colj = jt * 16 + ll;
        const float bj = bias ? bias[colj] : 0.0f;
#pragma unroll
        for (int v = 0; v < 8; ++v) {
            const int m = m0 + v + (lh ? 8 : 0);
            O[(size_t)m * 64 + colj] = c[jt][v] + bj;
        }
    }
}

// ---------------------------------------------------------------------------
// Row-wise l2norm over 64 columns (optionally preceded by ELU). Wave per row.
// ---------------------------------------------------------------------------
__global__ __launch_bounds__(256)
void rows_norm(const float* __restrict__ in, int R, float* __restrict__ out, int do_elu)
{
    const int lane = threadIdx.x & 31;
    const int row  = blockIdx.x * 8 + (threadIdx.x >> 5);
    if (row >= R) return;
    float v0 = in[(size_t)row * 64 + lane];
    float v1 = in[(size_t)row * 64 + lane + 32];
    if (do_elu) {
        v0 = v0 > 0.0f ? v0 : (expf(v0) - 1.0f);
        v1 = v1 > 0.0f ? v1 : (expf(v1) - 1.0f);
    }
    float ss    = wave_reduce_sum(v0 * v0 + v1 * v1);
    float scale = 1.0f / fmaxf(sqrtf(ss), 1e-12f);
    out[(size_t)row * 64 + lane]      = v0 * scale;
    out[(size_t)row * 64 + lane + 32] = v1 * scale;
}

// out = l2norm(a + b) per row. Wave per row.
__global__ __launch_bounds__(256)
void add_norm(const float* __restrict__ a, const float* __restrict__ b,
              int R, float* __restrict__ out)
{
    const int lane = threadIdx.x & 31;
    const int row  = blockIdx.x * 8 + (threadIdx.x >> 5);
    if (row >= R) return;
    float v0 = a[(size_t)row * 64 + lane]      + b[(size_t)row * 64 + lane];
    float v1 = a[(size_t)row * 64 + lane + 32] + b[(size_t)row * 64 + lane + 32];
    float ss    = wave_reduce_sum(v0 * v0 + v1 * v1);
    float scale = 1.0f / fmaxf(sqrtf(ss), 1e-12f);
    out[(size_t)row * 64 + lane]      = v0 * scale;
    out[(size_t)row * 64 + lane + 32] = v1 * scale;
}

// ---------------------------------------------------------------------------
// Attention score projection: S[n, soffs+h] = dot(att[h, :OUTsz], P[n, h*OUTsz:])
// ---------------------------------------------------------------------------
__global__ __launch_bounds__(256)
void score_k(const float* __restrict__ P, int R, const float* __restrict__ att,
             int H, int OUTsz, float* __restrict__ S, int sstride, int soffs)
{
    const int n = blockIdx.x * blockDim.x + threadIdx.x;
    if (n >= R) return;
    for (int h = 0; h < H; ++h) {
        const float* p = P + (size_t)n * 64 + h * OUTsz;
        const float* a = att + h * OUTsz;
        float s = 0.0f;
        for (int k = 0; k < OUTsz; ++k) s += a[k] * p[k];
        S[(size_t)n * sstride + soffs + h] = s;
    }
}

// ---------------------------------------------------------------------------
// Edge pass 1: e = exp(-leaky_relu(score, 0.2)); accumulate softmax denoms.
// sN row layout: [srI(H) | scI(H) | srO(H) | scO(H)]; sM row: [sgI(H) | sgO(H)].
// ---------------------------------------------------------------------------
__global__ __launch_bounds__(256)
void edge_pass1(const int* __restrict__ row, const int* __restrict__ col,
                const int* __restrict__ rel, int E, int H,
                const float* __restrict__ sN, const float* __restrict__ sM,
                float* __restrict__ eBuf,
                float* __restrict__ denI, float* __restrict__ denO)
{
    const int e = blockIdx.x * blockDim.x + threadIdx.x;
    if (e >= E) return;
    const int r = row[e], c = col[e], k = rel[e];
    const int ns = 4 * H, ms = 2 * H;
    for (int h = 0; h < H; ++h) {
        float aI = sN[(size_t)r * ns + h] + sN[(size_t)c * ns + H + h] + sM[(size_t)k * ms + h];
        float lI = aI > 0.0f ? aI : 0.2f * aI;
        float eI = expf(-lI);
        eBuf[(size_t)e * 2 * H + h] = eI;
        DEV_ATOMIC_ADD(&denI[(size_t)c * H + h], eI);

        float aO = sN[(size_t)r * ns + 2 * H + h] + sN[(size_t)c * ns + 3 * H + h] + sM[(size_t)k * ms + H + h];
        float lO = aO > 0.0f ? aO : 0.2f * aO;
        float eO = expf(-lO);
        eBuf[(size_t)e * 2 * H + H + h] = eO;
        DEV_ATOMIC_ADD(&denO[(size_t)r * H + h], eO);
    }
}

// ---------------------------------------------------------------------------
// Edge pass 2: gather c = Pr[row]+Pc[col]+Pg[rel]; scatter alpha*c.
// One wave per edge; lane handles columns {lane, lane+32}. Tables are L2-resident.
// ---------------------------------------------------------------------------
__global__ __launch_bounds__(256)
void edge_pass2(const int* __restrict__ row, const int* __restrict__ col,
                const int* __restrict__ rel, int E, int H,
                const float* __restrict__ PrI, const float* __restrict__ PcI,
                const float* __restrict__ PgI,
                const float* __restrict__ PrO, const float* __restrict__ PcO,
                const float* __restrict__ PgO,
                const float* __restrict__ eBuf,
                const float* __restrict__ denI, const float* __restrict__ denO,
                float* __restrict__ hI, float* __restrict__ hO)
{
    const int lane = threadIdx.x & 31;
    const int e    = blockIdx.x * 8 + (threadIdx.x >> 5);
    if (e >= E) return;
    const int r = row[e], c = col[e], k = rel[e];
#pragma unroll
    for (int half = 0; half < 2; ++half) {
        const int j = lane + 32 * half;
        const int h = (j * H) >> 6;   // head index: H=2 -> j/32, H=1 -> 0
        float cI = PrI[(size_t)r * 64 + j] + PcI[(size_t)c * 64 + j] + PgI[(size_t)k * 64 + j];
        float aI = eBuf[(size_t)e * 2 * H + h] / denI[(size_t)c * H + h];
        DEV_ATOMIC_ADD(&hI[(size_t)c * 64 + j], aI * cI);

        float cO = PrO[(size_t)r * 64 + j] + PcO[(size_t)c * 64 + j] + PgO[(size_t)k * 64 + j];
        float aO = eBuf[(size_t)e * 2 * H + H + h] / denO[(size_t)r * H + h];
        DEV_ATOMIC_ADD(&hO[(size_t)r * 64 + j], aO * cO);
    }
}

// ---------------------------------------------------------------------------
// Gated merge: lam = sigmoid([hi|ho] @ Wl.T + bl); out = lam*hi + (1-lam)*ho.
// Wave per row.
// ---------------------------------------------------------------------------
__global__ __launch_bounds__(256)
void merge_k(const float* __restrict__ hi, const float* __restrict__ ho,
             const float* __restrict__ Wl, const float* __restrict__ bl,
             int R, float* __restrict__ out)
{
    const int lane = threadIdx.x & 31;
    const int row  = blockIdx.x * 8 + (threadIdx.x >> 5);
    if (row >= R) return;
    float hi0 = hi[(size_t)row * 64 + lane];
    float hi1 = hi[(size_t)row * 64 + lane + 32];
    float ho0 = ho[(size_t)row * 64 + lane];
    float ho1 = ho[(size_t)row * 64 + lane + 32];
    float part = Wl[lane] * hi0 + Wl[lane + 32] * hi1
               + Wl[64 + lane] * ho0 + Wl[96 + lane] * ho1;
    float logit = wave_reduce_sum(part) + bl[0];
    float lam   = 1.0f / (1.0f + expf(-logit));
    out[(size_t)row * 64 + lane]      = lam * hi0 + (1.0f - lam) * ho0;
    out[(size_t)row * 64 + lane + 32] = lam * hi1 + (1.0f - lam) * ho1;
}

// ---------------------------------------------------------------------------
// Host-side launcher.
// ---------------------------------------------------------------------------
extern "C" void kernel_launch(void* const* d_in, const int* in_sizes, int n_in,
                              void* d_out, int out_size, void* d_ws, size_t ws_size,
                              hipStream_t stream)
{
    const float* x       = (const float*)d_in[0];
    const float* g       = (const float*)d_in[1];
    const float* W1_in1  = (const float*)d_in[2];
    const float* att_in1 = (const float*)d_in[3];
    const float* W1_out1 = (const float*)d_in[4];
    const float* att_out1= (const float*)d_in[5];
    const float* Wmi_i   = (const float*)d_in[6];
    const float* bmi_i   = (const float*)d_in[7];
    const float* Wmi_o   = (const float*)d_in[8];
    const float* bmi_o   = (const float*)d_in[9];
    const float* Wmi_l   = (const float*)d_in[10];
    const float* bmi_l   = (const float*)d_in[11];
    const float* Wrel    = (const float*)d_in[12];
    const float* W1_in2  = (const float*)d_in[13];
    const float* att_in2 = (const float*)d_in[14];
    const float* W1_out2 = (const float*)d_in[15];
    const float* att_out2= (const float*)d_in[16];
    const float* Wmo_i   = (const float*)d_in[17];
    const float* bmo_i   = (const float*)d_in[18];
    const float* Wmo_o   = (const float*)d_in[19];
    const float* bmo_o   = (const float*)d_in[20];
    const float* Wmo_l   = (const float*)d_in[21];
    const float* bmo_l   = (const float*)d_in[22];
    const float* Went    = (const float*)d_in[23];
    const int*   row     = (const int*)d_in[24];
    const int*   col     = (const int*)d_in[25];
    const int*   rel     = (const int*)d_in[26];

    const int N = in_sizes[0] / 64;
    const int M = in_sizes[1] / 64;
    const int E = in_sizes[24];

    float* ws = (float*)d_ws;
    size_t off = 0;
    auto alloc = [&](size_t n) { float* p = ws + off; off += n; return p; };
    float* xn   = alloc((size_t)N * 64);
    float* gn   = alloc((size_t)M * 64);
    float* PA   = alloc((size_t)N * 64);   // PrI / hInN / hIn2N
    float* PB   = alloc((size_t)N * 64);   // PcI / hOutN / hOut2N
    float* PC   = alloc((size_t)N * 64);   // PrO / hi
    float* PD   = alloc((size_t)N * 64);   // PcO / ho
    float* QgI  = alloc((size_t)M * 64);
    float* QgO  = alloc((size_t)M * 64);
    float* sN   = alloc((size_t)N * 8);
    float* sM   = alloc((size_t)M * 4);
    float* eBuf = alloc((size_t)E * 4);
    float* denI = alloc((size_t)N * 2);
    float* denO = alloc((size_t)N * 2);    // contiguous with denI
    float* aggI = alloc((size_t)N * 64);
    float* aggO = alloc((size_t)N * 64);   // contiguous with aggI
    float* hbuf = alloc((size_t)N * 64);   // h (layer1) then h2 (layer2)
    float* gp   = alloc((size_t)M * 64);   // g_prime (unnormalized)

    const dim3 B256(256);
    const dim3 rowsN((N + 7) / 8), rowsM((M + 7) / 8);
    const dim3 gemmN((N + 127) / 128), gemmM((M + 127) / 128);
    const dim3 thrN((N + 255) / 256), thrM((M + 255) / 256);
    const dim3 edge1((E + 255) / 256), edge2((E + 7) / 8);

    // ---------------- Input normalization ----------------
    rows_norm<<<rowsN, B256, 0, stream>>>(x, N, xn, 0);
    rows_norm<<<rowsM, B256, 0, stream>>>(g, M, gn, 0);

    // ---------------- Layer 1: projections (WMMA) ----------------
    gemm_x64<<<gemmN, B256, 0, stream>>>(xn, N, W1_in1, 192, 0,   nullptr, PA);
    gemm_x64<<<gemmN, B256, 0, stream>>>(xn, N, W1_in1, 192, 64,  nullptr, PB);
    gemm_x64<<<gemmM, B256, 0, stream>>>(gn, M, W1_in1, 192, 128, nullptr, QgI);
    gemm_x64<<<gemmN, B256, 0, stream>>>(xn, N, W1_out1, 192, 0,   nullptr, PC);
    gemm_x64<<<gemmN, B256, 0, stream>>>(xn, N, W1_out1, 192, 64,  nullptr, PD);
    gemm_x64<<<gemmM, B256, 0, stream>>>(gn, M, W1_out1, 192, 128, nullptr, QgO);

    // Scores (linearity of attention over decomposed projections)
    score_k<<<thrN, B256, 0, stream>>>(PA, N, att_in1,  2, 32, sN, 8, 0);
    score_k<<<thrN, B256, 0, stream>>>(PB, N, att_in1,  2, 32, sN, 8, 2);
    score_k<<<thrN, B256, 0, stream>>>(PC, N, att_out1, 2, 32, sN, 8, 4);
    score_k<<<thrN, B256, 0, stream>>>(PD, N, att_out1, 2, 32, sN, 8, 6);
    score_k<<<thrM, B256, 0, stream>>>(QgI, M, att_in1,  2, 32, sM, 4, 0);
    score_k<<<thrM, B256, 0, stream>>>(QgO, M, att_out1, 2, 32, sM, 4, 2);

    // Zero accumulators (capture-safe)
    hipMemsetAsync(denI, 0, (size_t)N * 4 * sizeof(float), stream);
    hipMemsetAsync(aggI, 0, (size_t)N * 128 * sizeof(float), stream);

    edge_pass1<<<edge1, B256, 0, stream>>>(row, col, rel, E, 2, sN, sM, eBuf, denI, denO);
    edge_pass2<<<edge2, B256, 0, stream>>>(row, col, rel, E, 2,
                                           PA, PB, QgI, PC, PD, QgO,
                                           eBuf, denI, denO, aggI, aggO);

    // elu + l2norm, then merge
    rows_norm<<<rowsN, B256, 0, stream>>>(aggI, N, PA, 1);
    rows_norm<<<rowsN, B256, 0, stream>>>(aggO, N, PB, 1);
    gemm_x64<<<gemmN, B256, 0, stream>>>(PA, N, Wmi_i, 64, 0, bmi_i, PC);
    gemm_x64<<<gemmN, B256, 0, stream>>>(PB, N, Wmi_o, 64, 0, bmi_o, PD);
    merge_k<<<rowsN, B256, 0, stream>>>(PC, PD, Wmi_l, bmi_l, N, hbuf);

    // g_prime = gn @ Wrel.T
    gemm_x64<<<gemmM, B256, 0, stream>>>(gn, M, Wrel, 64, 0, nullptr, gp);

    // ---------------- Layer 2 (heads=1, out=64) ----------------
    gemm_x64<<<gemmN, B256, 0, stream>>>(hbuf, N, W1_in2, 192, 0,   nullptr, PA);
    gemm_x64<<<gemmN, B256, 0, stream>>>(hbuf, N, W1_in2, 192, 64,  nullptr, PB);
    gemm_x64<<<gemmM, B256, 0, stream>>>(gp,   M, W1_in2, 192, 128, nullptr, QgI);
    gemm_x64<<<gemmN, B256, 0, stream>>>(hbuf, N, W1_out2, 192, 0,   nullptr, PC);
    gemm_x64<<<gemmN, B256, 0, stream>>>(hbuf, N, W1_out2, 192, 64,  nullptr, PD);
    gemm_x64<<<gemmM, B256, 0, stream>>>(gp,   M, W1_out2, 192, 128, nullptr, QgO);

    score_k<<<thrN, B256, 0, stream>>>(PA, N, att_in2,  1, 64, sN, 4, 0);
    score_k<<<thrN, B256, 0, stream>>>(PB, N, att_in2,  1, 64, sN, 4, 1);
    score_k<<<thrN, B256, 0, stream>>>(PC, N, att_out2, 1, 64, sN, 4, 2);
    score_k<<<thrN, B256, 0, stream>>>(PD, N, att_out2, 1, 64, sN, 4, 3);
    score_k<<<thrM, B256, 0, stream>>>(QgI, M, att_in2,  1, 64, sM, 2, 0);
    score_k<<<thrM, B256, 0, stream>>>(QgO, M, att_out2, 1, 64, sM, 2, 1);

    hipMemsetAsync(denI, 0, (size_t)N * 4 * sizeof(float), stream);
    hipMemsetAsync(aggI, 0, (size_t)N * 128 * sizeof(float), stream);

    edge_pass1<<<edge1, B256, 0, stream>>>(row, col, rel, E, 1, sN, sM, eBuf, denI, denO);
    edge_pass2<<<edge2, B256, 0, stream>>>(row, col, rel, E, 1,
                                           PA, PB, QgI, PC, PD, QgO,
                                           eBuf, denI, denO, aggI, aggO);

    rows_norm<<<rowsN, B256, 0, stream>>>(aggI, N, PA, 1);
    rows_norm<<<rowsN, B256, 0, stream>>>(aggO, N, PB, 1);
    gemm_x64<<<gemmN, B256, 0, stream>>>(PA, N, Wmo_i, 64, 0, bmo_i, PC);
    gemm_x64<<<gemmN, B256, 0, stream>>>(PB, N, Wmo_o, 64, 0, bmo_o, PD);
    merge_k<<<rowsN, B256, 0, stream>>>(PC, PD, Wmo_l, bmo_l, N, hbuf);   // h2

    // ---------------- Entity layer + outputs ----------------
    gemm_x64<<<gemmN, B256, 0, stream>>>(xn, N, Went, 64, 0, nullptr, aggI); // ent
    float* outH = (float*)d_out;
    float* outG = outH + (size_t)N * 64;
    add_norm<<<rowsN, B256, 0, stream>>>(aggI, hbuf, N, outH);
    rows_norm<<<rowsM, B256, 0, stream>>>(gp, M, outG, 0);
}
